// MLP_84207128805579
// MI455X (gfx1250) — 8-bit matrix operand order repaired and checked by reading, compiled, not run on hardware
//
#include <hip/hip_runtime.h>
#include <math.h>

typedef int v8i __attribute__((ext_vector_type(8)));

#define EMBED 768
#define HIDDEN 3072
#define TOKENS 32768
#define TOK 16
#define KC1 12
#define NT1 192
#define KC2 48
#define NT2 48

#define H_STRIDE 3088
#define XQ_STRIDE 784
#define OFF_HQ   197632
#define OFF_XQ   247040
#define OFF_RED  259584
#define OFF_A1   260608
#define SMEM_BYTES 260800

__global__ __launch_bounds__(256) void absmean_partial(
    const float* __restrict__ W, int n, float* __restrict__ partial) {
  __shared__ float red[256];
  float s = 0.f;
  for (int i = blockIdx.x * 256 + threadIdx.x; i < n; i += gridDim.x * 256)
    s += fabsf(W[i]);
  red[threadIdx.x] = s;
  __syncthreads();
  for (int off = 128; off > 0; off >>= 1) {
    if (threadIdx.x < off) red[threadIdx.x] += red[threadIdx.x + off];
    __syncthreads();
  }
  if (threadIdx.x == 0) partial[blockIdx.x] = red[0];
}

__global__ void finalize_scales(const float* __restrict__ p1,
                                const float* __restrict__ p2,
                                float* __restrict__ sc, int n1, int n2) {
  if (threadIdx.x == 0 && blockIdx.x == 0) {
    float s1 = 0.f, s2 = 0.f;
    for (int i = 0; i < 256; ++i) s1 += p1[i];
    for (int i = 0; i < 256; ++i) s2 += p2[i];
    float m1 = fmaxf(s1 / (float)n1, 1e-5f);
    float m2 = fmaxf(s2 / (float)n2, 1e-5f);
    sc[0] = 1.f / m1;  sc[1] = m1;
    sc[2] = 1.f / m2;  sc[3] = m2;
  }
}

__global__ __launch_bounds__(256) void pack_weights(
    const float* __restrict__ W, const float* __restrict__ sc, int sidx,
    unsigned* __restrict__ Wp, int F, int D) {
  const float s = sc[sidx];
  const int KC = D >> 6;
  const int total = (F >> 4) * KC * 256;
  for (int idx = blockIdx.x * 256 + threadIdx.x; idx < total;
       idx += gridDim.x * 256) {
    const int v = idx & 7;
    const int lane = (idx >> 3) & 31;
    const int chunk = idx >> 8;
    const int kc = chunk % KC;
    const int tf = chunk / KC;
    const int n = lane & 15, half = lane >> 4;
    const int f = tf * 16 + n;
    const int kbase = kc * 64 + 16 * (v >> 1) + 8 * half + 4 * (v & 1);
    unsigned word = 0;
#pragma unroll
    for (int b = 0; b < 4; ++b) {
      float w = W[(long)f * D + kbase + b];
      int q = (int)rintf(w * s);
      q = q < -1 ? -1 : (q > 1 ? 1 : q);
      word |= ((unsigned)(q & 0xFF)) << (8 * b);
    }
    Wp[idx] = word;
  }
}

__global__ __launch_bounds__(256, 1) void bitff_fused(
    const float* __restrict__ x, const float* __restrict__ g1,
    const float* __restrict__ g2, const unsigned* __restrict__ w1p,
    const unsigned* __restrict__ w2p, const float* __restrict__ sc,
    float* __restrict__ out) {
  extern __shared__ char smem[];
  float* hbuf = (float*)smem;
  char* hq = smem + OFF_HQ;
  char* xq = smem + OFF_XQ;
  float* red = (float*)(smem + OFF_RED);
  float* a1 = (float*)(smem + OFF_A1);
  float* a2 = a1 + 16;
  float* rr = a2 + 16;

  const int tid = threadIdx.x;
  const int lane = tid & 31;
  const int wave = tid >> 5;
  const int m = lane & 15;
  const int halfk = lane >> 4;
  const long t0 = (long)blockIdx.x * TOK;
  const float dq_w1 = sc[1], dq_w2 = sc[3];
  const int t = tid >> 4, th = tid & 15;

  {
    const float* xrow = x + (t0 + t) * EMBED;
    float* srow = hbuf + t * H_STRIDE;
    float ss = 0.f;
    for (int c = th; c < EMBED; c += 16) { float v = xrow[c]; srow[c] = v; ss += v * v; }
    red[t * 16 + th] = ss;
  }
  __syncthreads();
  if (tid < TOK) {
    float s = 0.f;
    for (int i = 0; i < 16; ++i) s += red[tid * 16 + i];
    rr[tid] = rsqrtf(s * (1.f / EMBED) + 1e-6f);
  }
  __syncthreads();
  {
    const float* srow = hbuf + t * H_STRIDE;
    const float r = rr[t];
    float mx = 0.f;
    for (int c = th; c < EMBED; c += 16) mx = fmaxf(mx, fabsf(srow[c] * r * g1[c]));
    red[t * 16 + th] = mx;
  }
  __syncthreads();
  if (tid < TOK) {
    float mx = 0.f;
    for (int i = 0; i < 16; ++i) mx = fmaxf(mx, red[tid * 16 + i]);
    mx = fmaxf(mx, 1e-5f);
    a1[tid] = (mx * (1.f / 127.f)) * dq_w1;
    rr[tid] = rr[tid] * (127.f / mx);
  }
  __syncthreads();
  {
    const float* srow = hbuf + t * H_STRIDE;
    const float q = rr[t];
    signed char* qrow = (signed char*)(xq + t * XQ_STRIDE);
    for (int c = th; c < EMBED; c += 16) {
      float v = rintf(srow[c] * q * g1[c]);
      v = fminf(fmaxf(v, -128.f), 127.f);
      qrow[c] = (signed char)(int)v;
    }
  }
  __syncthreads();

  v8i afrag[KC1];
  {
    const char* arow = xq + m * XQ_STRIDE + 8 * halfk;
#pragma unroll
    for (int kc = 0; kc < KC1; ++kc) {
      const char* p = arow + kc * 64;
      int2 d01 = *(const int2*)(p);
      int2 d23 = *(const int2*)(p + 16);
      int2 d45 = *(const int2*)(p + 32);
      int2 d67 = *(const int2*)(p + 48);
      v8i a;
      a[0] = d01.x; a[1] = d01.y; a[2] = d23.x; a[3] = d23.y;
      a[4] = d45.x; a[5] = d45.y; a[6] = d67.x; a[7] = d67.y;
      afrag[kc] = a;
    }
  }
  for (int tf = wave; tf < NT1; tf += 8) {
    v8i acc = {0, 0, 0, 0, 0, 0, 0, 0};
    const unsigned* bp = w1p + (long)tf * KC1 * 256 + lane * 8;
#pragma unroll
    for (int kc = 0; kc < KC1; ++kc) {
      int4 b0 = *(const int4*)(bp);
      int4 b1 = *(const int4*)(bp + 4);
      v8i b;
      b[0] = b0.x; b[1] = b0.y; b[2] = b0.z; b[3] = b0.w;
      b[4] = b1.x; b[5] = b1.y; b[6] = b1.z; b[7] = b1.w;
      acc = __builtin_amdgcn_wmma_i32_16x16x64_iu8(true, afrag[kc], true, b,
                                                   acc, false, false);
      bp += 256;
    }
    const int mrow = 8 * halfk;
#pragma unroll
    for (int r = 0; r < 8; ++r) {
      float h = (float)acc[r] * a1[r + mrow];
      float gel = 0.5f * h * (1.f + erff(h * 0.70710678118654752f));
      hbuf[(r + mrow) * H_STRIDE + tf * 16 + m] = gel;
    }
  }
  __syncthreads();

  {
    const float* srow = hbuf + t * H_STRIDE;
    float ss = 0.f;
    for (int c = th; c < HIDDEN; c += 16) { float v = srow[c]; ss += v * v; }
    red[t * 16 + th] = ss;
  }
  __syncthreads();
  if (tid < TOK) {
    float s = 0.f;
    for (int i = 0; i < 16; ++i) s += red[tid * 16 + i];
    rr[tid] = rsqrtf(s * (1.f / HIDDEN) + 1e-6f);
  }
  __syncthreads();
  {
    const float* srow = hbuf + t * H_STRIDE;
    const float r = rr[t];
    float mx = 0.f;
    for (int c = th; c < HIDDEN; c += 16) mx = fmaxf(mx, fabsf(srow[c] * r * g2[c]));
    red[t * 16 + th] = mx;
  }
  __syncthreads();
  if (tid < TOK) {
    float mx = 0.f;
    for (int i = 0; i < 16; ++i) mx = fmaxf(mx, red[tid * 16 + i]);
    mx = fmaxf(mx, 1e-5f);
    a2[tid] = (mx * (1.f / 127.f)) * dq_w2;
    rr[tid] = rr[tid] * (127.f / mx);
  }
  __syncthreads();
  {
    const float* srow = hbuf + t * H_STRIDE;
    const float q = rr[t];
    signed char* qrow = (signed char*)(hq + t * H_STRIDE);
    for (int c = th; c < HIDDEN; c += 16) {
      float v = rintf(srow[c] * q * g2[c]);
      v = fminf(fmaxf(v, -128.f), 127.f);
      qrow[c] = (signed char)(int)v;
    }
  }
  __syncthreads();

  for (int tf = wave; tf < NT2; tf += 8) {
    v8i acc = {0, 0, 0, 0, 0, 0, 0, 0};
    const unsigned* bp = w2p + (long)tf * KC2 * 256 + lane * 8;
    const char* arow = hq + m * H_STRIDE + 8 * halfk;
#pragma unroll 4
    for (int kc = 0; kc < KC2; ++kc) {
      const char* p = arow + kc * 64;
      int2 d01 = *(const int2*)(p);
      int2 d23 = *(const int2*)(p + 16);
      int2 d45 = *(const int2*)(p + 32);
      int2 d67 = *(const int2*)(p + 48);
      v8i a;
      a[0] = d01.x; a[1] = d01.y; a[2] = d23.x; a[3] = d23.y;
      a[4] = d45.x; a[5] = d45.y; a[6] = d67.x; a[7] = d67.y;
      int4 b0 = *(const int4*)(bp);
      int4 b1 = *(const int4*)(bp + 4);
      v8i b;
      b[0] = b0.x; b[1] = b0.y; b[2] = b0.z; b[3] = b0.w;
      b[4] = b1.x; b[5] = b1.y; b[6] = b1.z; b[7] = b1.w;
      acc = __builtin_amdgcn_wmma_i32_16x16x64_iu8(true, a, true, b, acc,
                                                   false, false);
      bp += 256;
    }
    const int mrow = 8 * halfk;
#pragma unroll
    for (int r = 0; r < 8; ++r) {
      out[(t0 + r + mrow) * EMBED + tf * 16 + m] = (float)acc[r] * a2[r + mrow];
    }
  }
}

extern "C" void kernel_launch(void* const* d_in, const int* in_sizes, int n_in,
                              void* d_out, int out_size, void* d_ws,
                              size_t ws_size, hipStream_t stream) {
  const float* x = (const float*)d_in[0];
  const float* w1 = (const float*)d_in[1];
  const float* g1 = (const float*)d_in[2];
  const float* w2 = (const float*)d_in[3];
  const float* g2 = (const float*)d_in[4];
  float* out = (float*)d_out;

  const int NW = HIDDEN * EMBED;
  unsigned* w1p = (unsigned*)d_ws;
  unsigned* w2p = w1p + (NW / 4);
  float* part1 = (float*)(w2p + (NW / 4));
  float* part2 = part1 + 256;
  float* sc = part2 + 256;

  absmean_partial<<<256, 256, 0, stream>>>(w1, NW, part1);
  absmean_partial<<<256, 256, 0, stream>>>(w2, NW, part2);
  finalize_scales<<<1, 64, 0, stream>>>(part1, part2, sc, NW, NW);
  pack_weights<<<1152, 256, 0, stream>>>(w1, sc, 0, w1p, HIDDEN, EMBED);
  pack_weights<<<1152, 256, 0, stream>>>(w2, sc, 2, w2p, EMBED, HIDDEN);
  bitff_fused<<<TOKENS / TOK, 256, SMEM_BYTES, stream>>>(x, g1, g2, w1p, w2p,
                                                         sc, out);
}
